// VAE_IGLS_35390530519136
// MI455X (gfx1250) — compile-verified, matrix operand order checked by reading, not run on hardware
//
#include <hip/hip_runtime.h>
#include <hip/hip_bf16.h>
#include <math.h>

typedef __attribute__((ext_vector_type(16))) _Float16 v16h;
typedef __attribute__((ext_vector_type(8)))  _Float16 v8h;
typedef __attribute__((ext_vector_type(8)))  float    v8f;

// ---------------------------------------------------------------------------
// Implicit-GEMM 3D convolution (forward or lhs-dilated transposed) via WMMA.
// One wave computes a 16(cout) x 16(position) tile. Fill-row r = lane>>1 is
// lane-constant, so all position decoding is hoisted out of the K loop and
// the k->(ci,kd,kh,kw) decomposition advances by carry increments.
// ---------------------------------------------------------------------------
__global__ __launch_bounds__(32)
void conv3d_wmma(const float* __restrict__ X, const float* __restrict__ W,
                 const float* __restrict__ bias, float* __restrict__ Y,
                 int B, int Cin, int Di, int Hi, int Wi,
                 int Cout, int Do, int Ho, int Wo,
                 int stride, int padlo, int transposed, int actmode)
{
  __shared__ __align__(16) _Float16 As[16 * 32];   // [m][k]
  __shared__ __align__(16) _Float16 Bs[16 * 32];   // [n][k]
  const int  lane  = threadIdx.x;
  const long Npos  = (long)B * Do * Ho * Wo;
  const long nBase = (long)blockIdx.x * 16;
  const int  mBase = blockIdx.y * 16;
  const int  Ktot  = Cin * 27;

  // ---- lane-constant fill-row state (hoisted out of K loop) ----
  const int  r    = lane >> 1;         // tile row this lane fills
  const int  kof  = (lane & 1) * 16;   // k offset inside the row
  const int  m    = mBase + r;
  const bool mok  = (m < Cout);

  const long HiWi   = (long)Hi * Wi;
  const long DiHiWi = (long)Di * HiWi;
  long bbase = 0; int d0 = 0, h0 = 0, w0 = 0;
  const long posF = nBase + r;
  const bool pok  = (posF < Npos);
  if (pok) {
    int ow = (int)(posF % Wo); long tpp = posF / Wo;
    int oh = (int)(tpp % Ho); tpp /= Ho;
    int od = (int)(tpp % Do);
    int bb = (int)(tpp / Do);
    bbase = (long)bb * Cin * DiHiWi;
    if (!transposed) { d0 = od * stride - padlo; h0 = oh * stride - padlo; w0 = ow * stride - padlo; }
    else             { d0 = od - padlo;          h0 = oh - padlo;          w0 = ow - padlo; }
  }
  __builtin_prefetch(X + bbase, 0, 1);   // global_prefetch_b8

  v8f acc = {};
  for (int k0 = 0; k0 < Ktot; k0 += 32) {
    // decompose once per K step, then carry-increment
    int k   = k0 + kof;
    int ci  = k / 27;
    int rem = k - ci * 27;
    int kd  = rem / 9;
    int r9  = rem - kd * 9;
    int kh  = r9 / 3;
    int kw  = r9 - kh * 3;

    __align__(16) _Float16 aseg[16];
    __align__(16) _Float16 bseg[16];
#pragma unroll
    for (int i = 0; i < 16; ++i) {
      float av = 0.f, bv = 0.f;
      if (k < Ktot) {
        if (mok) {
          long widx = transposed
            ? (long)ci * (Cout * 27) + (long)m * 27 + (26 - (kd * 9 + kh * 3 + kw))
            : (long)m * Ktot + k;
          av = W[widx];
        }
        if (pok) {
          int id, ih, iw; bool ok;
          if (!transposed) {
            id = d0 + kd; ih = h0 + kh; iw = w0 + kw;
            ok = (id >= 0 && id < Di && ih >= 0 && ih < Hi && iw >= 0 && iw < Wi);
          } else {
            int jd = d0 + kd, jh = h0 + kh, jw = w0 + kw;
            ok = (jd >= 0 && jh >= 0 && jw >= 0 &&
                  (jd % stride) == 0 && (jh % stride) == 0 && (jw % stride) == 0);
            id = jd / stride; ih = jh / stride; iw = jw / stride;
            ok = ok && (id < Di) && (ih < Hi) && (iw < Wi);
          }
          if (ok) bv = X[bbase + (long)ci * DiHiWi + (long)id * HiWi + (long)ih * Wi + iw];
        }
      }
      aseg[i] = (_Float16)av;
      bseg[i] = (_Float16)bv;
      ++k;
      if (++kw == 3) { kw = 0; if (++kh == 3) { kh = 0; if (++kd == 3) { kd = 0; ++ci; } } }
    }
    // two ds_store_b128 per matrix per lane
    *(v8h*)&As[r * 32 + kof]     = *(const v8h*)&aseg[0];
    *(v8h*)&As[r * 32 + kof + 8] = *(const v8h*)&aseg[8];
    *(v8h*)&Bs[r * 32 + kof]     = *(const v8h*)&bseg[0];
    *(v8h*)&Bs[r * 32 + kof + 8] = *(const v8h*)&bseg[8];
    __syncthreads();

    // ---- consume per CDNA5 16-bit A/B layout ----
    const int row  = lane & 15;
    const int koff = (lane >> 4) * 8;   // lane<16: K0..7,16..23 ; lane>=16: K8..15,24..31
    v16h a, b;
    {
      v8h a0 = *(const v8h*)&As[row * 32 + koff];
      v8h a1 = *(const v8h*)&As[row * 32 + koff + 16];
      v8h b0 = *(const v8h*)&Bs[row * 32 + koff];
      v8h b1 = *(const v8h*)&Bs[row * 32 + koff + 16];
#pragma unroll
      for (int i = 0; i < 8; ++i) { a[i] = a0[i]; a[i + 8] = a1[i]; b[i] = b0[i]; b[i + 8] = b1[i]; }
    }
    acc = __builtin_amdgcn_wmma_f32_16x16x32_f16(false, a, false, b, (short)0, acc, false, false);
    __syncthreads();
  }

  // ---- epilogue: D layout: vgpr r2 -> M = r2 + (lane>=16 ? 8 : 0), N = lane&15
  {
    const int n    = lane & 15;
    const int mofs = (lane >> 4) * 8;
    long pos = nBase + n;
    if (pos < Npos) {
      int ow = (int)(pos % Wo); long tpp = pos / Wo;
      int oh = (int)(tpp % Ho); tpp /= Ho;
      int od = (int)(tpp % Do);
      int bb = (int)(tpp / Do);
      const long sp = ((long)od * Ho + oh) * Wo + ow;
      const long DoHoWo = (long)Do * Ho * Wo;
#pragma unroll
      for (int r2 = 0; r2 < 8; ++r2) {
        int mm = mBase + mofs + r2;
        if (mm < Cout) {
          float v = acc[r2] + bias[mm];
          if (actmode == 1) v = 1.f / (1.f + __expf(-v));
          Y[((long)bb * Cout + mm) * DoHoWo + sp] = v;
        }
      }
    }
  }
}

// ---------------------------------------------------------------------------
// GEMM (NT) via WMMA:  Y[n,m] = sum_k A[m,k]*B[n,k] + bias[m]
// A: [M,K] rm, B: [N,K] rm, Y: [N,M] rm.  Fully affine addressing.
// ---------------------------------------------------------------------------
__global__ __launch_bounds__(32)
void gemm_nt_wmma(const float* __restrict__ A, const float* __restrict__ Bm,
                  const float* __restrict__ bias, float* __restrict__ Yt,
                  int Mdim, int Ndim, int Kdim)
{
  __shared__ __align__(16) _Float16 As[16 * 32];
  __shared__ __align__(16) _Float16 Bs[16 * 32];
  const int lane  = threadIdx.x;
  const int nBase = blockIdx.x * 16;
  const int mBase = blockIdx.y * 16;

  const int  r   = lane >> 1;
  const int  kof = (lane & 1) * 16;
  const int  m   = mBase + r;
  const int  n   = nBase + r;
  const bool mok = (m < Mdim);
  const bool nok = (n < Ndim);
  const float* Arow = A  + (long)m * Kdim;
  const float* Brow = Bm + (long)n * Kdim;

  v8f acc = {};
  for (int k0 = 0; k0 < Kdim; k0 += 32) {
    __align__(16) _Float16 aseg[16];
    __align__(16) _Float16 bseg[16];
#pragma unroll
    for (int i = 0; i < 16; ++i) {
      int k = k0 + kof + i;
      aseg[i] = (_Float16)((mok && k < Kdim) ? Arow[k] : 0.f);
      bseg[i] = (_Float16)((nok && k < Kdim) ? Brow[k] : 0.f);
    }
    *(v8h*)&As[r * 32 + kof]     = *(const v8h*)&aseg[0];
    *(v8h*)&As[r * 32 + kof + 8] = *(const v8h*)&aseg[8];
    *(v8h*)&Bs[r * 32 + kof]     = *(const v8h*)&bseg[0];
    *(v8h*)&Bs[r * 32 + kof + 8] = *(const v8h*)&bseg[8];
    __syncthreads();

    const int row  = lane & 15;
    const int koff = (lane >> 4) * 8;
    v16h a, b;
    {
      v8h a0 = *(const v8h*)&As[row * 32 + koff];
      v8h a1 = *(const v8h*)&As[row * 32 + koff + 16];
      v8h b0 = *(const v8h*)&Bs[row * 32 + koff];
      v8h b1 = *(const v8h*)&Bs[row * 32 + koff + 16];
#pragma unroll
      for (int i = 0; i < 8; ++i) { a[i] = a0[i]; a[i + 8] = a1[i]; b[i] = b0[i]; b[i + 8] = b1[i]; }
    }
    acc = __builtin_amdgcn_wmma_f32_16x16x32_f16(false, a, false, b, (short)0, acc, false, false);
    __syncthreads();
  }
  {
    const int nn   = nBase + (lane & 15);
    const int mofs = (lane >> 4) * 8;
    if (nn < Ndim) {
#pragma unroll
      for (int r2 = 0; r2 < 8; ++r2) {
        int mm = mBase + mofs + r2;
        if (mm < Mdim) Yt[(long)nn * Mdim + mm] = acc[r2] + bias[mm];
      }
    }
  }
}

// ---------------------------------------------------------------------------
// BatchNorm3d training-mode batch stats (biased var), one block per channel.
// ---------------------------------------------------------------------------
__global__ __launch_bounds__(256)
void bn_stats(const float* __restrict__ X, float* __restrict__ meanvar,
              int B, int C, long S)
{
  const int c = blockIdx.x;
  float s = 0.f, sq = 0.f;
  const long tot = (long)B * S;
  for (long i = threadIdx.x; i < tot; i += 256) {
    long b = i / S, sp = i - b * S;
    float v = X[((long)b * C + c) * S + sp];
    s += v; sq += v * v;
  }
  __shared__ float rs[256], rq[256];
  rs[threadIdx.x] = s; rq[threadIdx.x] = sq;
  __syncthreads();
  for (int o = 128; o > 0; o >>= 1) {
    if (threadIdx.x < o) { rs[threadIdx.x] += rs[threadIdx.x + o]; rq[threadIdx.x] += rq[threadIdx.x + o]; }
    __syncthreads();
  }
  if (threadIdx.x == 0) {
    float mn = rs[0] / (float)tot;
    meanvar[c]     = mn;
    meanvar[C + c] = rq[0] / (float)tot - mn * mn;
  }
}

__global__ __launch_bounds__(256)
void bn_apply_lrelu(float* __restrict__ X, const float* __restrict__ meanvar,
                    const float* __restrict__ g, const float* __restrict__ be,
                    int B, int C, long S)
{
  const long tot = (long)B * C * S;
  for (long i = (long)blockIdx.x * blockDim.x + threadIdx.x; i < tot;
       i += (long)gridDim.x * blockDim.x) {
    int c = (int)((i / S) % C);
    float mn = meanvar[c], vr = meanvar[C + c];
    float y = g[c] * (X[i] - mn) * rsqrtf(vr + 1e-5f) + be[c];
    X[i] = (y >= 0.f) ? y : 0.01f * y;
  }
}

// ---------------------------------------------------------------------------
// Fused IGLS (10 iters) + 32x32 Gauss-Jordan inverse + cyclic Jacobi eigen +
// MVN sample. One block per latent dim k (columns fully independent).
// ---------------------------------------------------------------------------
__global__ __launch_bounds__(128)
void igls_eig(const float* __restrict__ Z, const int* __restrict__ sid,
              const float* __restrict__ tv, float* __restrict__ Sigma_out,
              float* __restrict__ beta_out, float* __restrict__ zhat_out, int K)
{
  const int k   = blockIdx.x;
  const int tid = threadIdx.x;
  __shared__ float zk[32], tt[32], rr[32], uu[32], vv[32], epsv[32], sqw[32];
  __shared__ int   ss[32];
  __shared__ float Sig[32][33];
  __shared__ float GJ[32][66];
  __shared__ float EV[32][33];
  __shared__ float G10[10];
  __shared__ float Ginv[16];
  __shared__ float s4[4], sgv[4];
  __shared__ float bcast[8];

  if (tid < 32) { zk[tid] = Z[(long)tid * K + k]; tt[tid] = tv[tid]; ss[tid] = sid[tid]; }
  if (tid < 10) G10[tid] = 0.f;
  __syncthreads();

  {
    float g[10] = {0,0,0,0,0,0,0,0,0,0};
    for (int p = tid; p < 1024; p += 128) {
      int i = p >> 5, j = p & 31;
      float d  = (i == j) ? 1.f : 0.f;
      float sm = (ss[i] == ss[j]) ? 1.f : 0.f;
      float z3 = sm * (tt[i] + tt[j]);
      float z4 = sm * (tt[i] * tt[j]);
      g[0] += d * d;   g[1] += d * sm;  g[2] += d * z3;  g[3] += d * z4;
      g[4] += sm * sm; g[5] += sm * z3; g[6] += sm * z4;
      g[7] += z3 * z3; g[8] += z3 * z4; g[9] += z4 * z4;
    }
    for (int e = 0; e < 10; ++e) atomicAdd(&G10[e], g[e]);
  }
  __syncthreads();
  if (tid == 0) {
    float A[4][8];
    float Gm[4][4] = {{G10[0],G10[1],G10[2],G10[3]},
                      {G10[1],G10[4],G10[5],G10[6]},
                      {G10[2],G10[5],G10[7],G10[8]},
                      {G10[3],G10[6],G10[8],G10[9]}};
    for (int i = 0; i < 4; ++i)
      for (int j = 0; j < 8; ++j) A[i][j] = (j < 4) ? Gm[i][j] : ((j - 4 == i) ? 1.f : 0.f);
    for (int c = 0; c < 4; ++c) {
      float ip = 1.f / A[c][c];
      for (int j = 0; j < 8; ++j) A[c][j] *= ip;
      for (int rr2 = 0; rr2 < 4; ++rr2) if (rr2 != c) {
        float f = A[rr2][c];
        for (int j = 0; j < 8; ++j) A[rr2][j] -= f * A[c][j];
      }
    }
    for (int i = 0; i < 4; ++i)
      for (int j = 0; j < 4; ++j) Ginv[i * 4 + j] = A[i][4 + j];
    float St = 0, St2 = 0, Sz = 0, Stz = 0;
    for (int i = 0; i < 32; ++i) { St += tt[i]; St2 += tt[i]*tt[i]; Sz += zk[i]; Stz += tt[i]*zk[i]; }
    float det = 32.f * St2 - St * St;
    bcast[0] = (St2 * Sz - St * Stz) / det;
    bcast[1] = (32.f * Stz - St * Sz) / det;
  }
  __syncthreads();

  for (int it = 0; it < 10; ++it) {
    if (tid < 32) rr[tid] = bcast[0] + bcast[1] * tt[tid] - zk[tid];
    if (tid < 4)  s4[tid] = 0.f;
    __syncthreads();
    {
      float l0 = 0, l1 = 0, l2 = 0, l3 = 0;
      for (int p = tid; p < 1024; p += 128) {
        int i = p >> 5, j = p & 31;
        float val = rr[i] * rr[j];
        if (i == j) l0 += val;
        if (ss[i] == ss[j]) { l1 += val; l2 += (tt[i] + tt[j]) * val; l3 += tt[i] * tt[j] * val; }
      }
      atomicAdd(&s4[0], l0); atomicAdd(&s4[1], l1);
      atomicAdd(&s4[2], l2); atomicAdd(&s4[3], l3);
    }
    __syncthreads();
    if (tid < 4)
      sgv[tid] = Ginv[tid*4+0]*s4[0] + Ginv[tid*4+1]*s4[1] + Ginv[tid*4+2]*s4[2] + Ginv[tid*4+3]*s4[3];
    __syncthreads();
    for (int p = tid; p < 1024; p += 128) {
      int i = p >> 5, j = p & 31;
      float sm = (ss[i] == ss[j]) ? 1.f : 0.f;
      float v = sgv[0] * ((i == j) ? 1.f : 0.f)
              + sm * (sgv[1] + sgv[2] * (tt[i] + tt[j]) + sgv[3] * tt[i] * tt[j]);
      Sig[i][j] = v;
      GJ[i][j] = v;
      GJ[i][32 + j] = (i == j) ? 1.f : 0.f;
    }
    __syncthreads();
    for (int c = 0; c < 32; ++c) {
      if (tid == 0) bcast[2] = 1.f / GJ[c][c];
      __syncthreads();
      if (tid < 64) GJ[c][tid] *= bcast[2];
      __syncthreads();
      if (tid < 32) uu[tid] = GJ[tid][c];
      __syncthreads();
      for (int e = tid; e < 2048; e += 128) {
        int r2 = e >> 6, c2 = e & 63;
        if (r2 != c) GJ[r2][c2] -= uu[r2] * GJ[c][c2];
      }
      __syncthreads();
    }
    if (tid < 32) {
      float su = 0.f, sv = 0.f;
      for (int i = 0; i < 32; ++i) { float w = GJ[i][32 + tid]; su += w; sv += tt[i] * w; }
      uu[tid] = su; vv[tid] = sv;
    }
    __syncthreads();
    if (tid == 0) {
      float A00 = 0, A01 = 0, A11 = 0, r0 = 0, r1 = 0;
      for (int j = 0; j < 32; ++j) {
        A00 += uu[j]; A01 += uu[j] * tt[j]; A11 += vv[j] * tt[j];
        r0 += uu[j] * zk[j]; r1 += vv[j] * zk[j];
      }
      float det = A00 * A11 - A01 * A01;
      bcast[0] = ( A11 * r0 - A01 * r1) / det;
      bcast[1] = (-A01 * r0 + A00 * r1) / det;
    }
    __syncthreads();
  }

  if (tid == 0) { beta_out[k] = bcast[0]; beta_out[K + k] = bcast[1]; }
  for (int p = tid; p < 1024; p += 128)
    Sigma_out[(long)k * 1024 + p] = Sig[p >> 5][p & 31];
  for (int p = tid; p < 1024; p += 128)
    EV[p >> 5][p & 31] = ((p >> 5) == (p & 31)) ? 1.f : 0.f;
  __syncthreads();

  for (int sweep = 0; sweep < 6; ++sweep) {
    for (int pp = 0; pp < 31; ++pp) {
      for (int qq = pp + 1; qq < 32; ++qq) {
        if (tid == 0) {
          float apq = Sig[pp][qq];
          float c = 1.f, s = 0.f;
          if (fabsf(apq) > 1e-12f) {
            float theta = 0.5f * (Sig[qq][qq] - Sig[pp][pp]) / apq;
            float tj = ((theta >= 0.f) ? 1.f : -1.f) / (fabsf(theta) + sqrtf(theta * theta + 1.f));
            c = rsqrtf(tj * tj + 1.f); s = tj * c;
          }
          bcast[3] = c; bcast[4] = s;
        }
        __syncthreads();
        float c = bcast[3], s = bcast[4];
        if (tid < 32) {
          float a1 = Sig[pp][tid], a2 = Sig[qq][tid];
          Sig[pp][tid] = c * a1 - s * a2;
          Sig[qq][tid] = s * a1 + c * a2;
        }
        __syncthreads();
        if (tid < 32) {
          float a1 = Sig[tid][pp], a2 = Sig[tid][qq];
          Sig[tid][pp] = c * a1 - s * a2;
          Sig[tid][qq] = s * a1 + c * a2;
          float v1 = EV[tid][pp], v2 = EV[tid][qq];
          EV[tid][pp] = c * v1 - s * v2;
          EV[tid][qq] = s * v1 + c * v2;
        }
        __syncthreads();
      }
    }
  }
  if (tid < 32) {
    float w = Sig[tid][tid]; if (w < 1e-6f) w = 1e-6f;
    sqw[tid] = sqrtf(w);
    unsigned s0 = (unsigned)(k * 32 + tid) * 747796405u + 2891336453u;
    unsigned w1 = ((s0 >> ((s0 >> 28) + 4u)) ^ s0) * 277803737u; w1 = (w1 >> 22) ^ w1;
    unsigned s1 = s0 * 747796405u + 2891336453u;
    unsigned w2 = ((s1 >> ((s1 >> 28) + 4u)) ^ s1) * 277803737u; w2 = (w2 >> 22) ^ w2;
    float u1 = ((w1 >> 8) + 1u) * (1.f / 16777216.f);
    float u2 = (w2 >> 8) * (1.f / 16777216.f);
    epsv[tid] = sqrtf(-2.f * logf(u1)) * __cosf(6.28318530718f * u2);
  }
  __syncthreads();
  if (tid < 32) {
    float acc = 0.f;
    for (int c2 = 0; c2 < 32; ++c2) acc += EV[tid][c2] * sqw[c2] * epsv[c2];
    zhat_out[(long)tid * K + k] = acc;
  }
}

// ---------------------------------------------------------------------------
extern "C" void kernel_launch(void* const* d_in, const int* in_sizes, int n_in,
                              void* d_out, int out_size, void* d_ws, size_t ws_size,
                              hipStream_t stream)
{
  (void)in_sizes; (void)n_in; (void)out_size; (void)ws_size;
  const float* x      = (const float*)d_in[0];
  const float* enc_w1 = (const float*)d_in[1];
  const float* enc_b1 = (const float*)d_in[2];
  const float* enc_g1 = (const float*)d_in[3];
  const float* enc_e1 = (const float*)d_in[4];
  const float* enc_w2 = (const float*)d_in[5];
  const float* enc_b2 = (const float*)d_in[6];
  const float* enc_g2 = (const float*)d_in[7];
  const float* enc_e2 = (const float*)d_in[8];
  const float* enc_w3 = (const float*)d_in[9];
  const float* enc_b3 = (const float*)d_in[10];
  const float* enc_g3 = (const float*)d_in[11];
  const float* enc_e3 = (const float*)d_in[12];
  const float* enc_w4 = (const float*)d_in[13];
  const float* enc_b4 = (const float*)d_in[14];
  const float* enc_g4 = (const float*)d_in[15];
  const float* enc_e4 = (const float*)d_in[16];
  const float* mu_w   = (const float*)d_in[17];
  const float* mu_b   = (const float*)d_in[18];
  const float* dec_lw = (const float*)d_in[19];
  const float* dec_lb = (const float*)d_in[20];
  const float* dec_w1 = (const float*)d_in[21];
  const float* dec_b1 = (const float*)d_in[22];
  const float* dec_g1 = (const float*)d_in[23];
  const float* dec_e1 = (const float*)d_in[24];
  const float* dec_w2 = (const float*)d_in[25];
  const float* dec_b2 = (const float*)d_in[26];
  const float* dec_g2 = (const float*)d_in[27];
  const float* dec_e2 = (const float*)d_in[28];
  const float* dec_w3 = (const float*)d_in[29];
  const float* dec_b3 = (const float*)d_in[30];
  const float* dec_g3 = (const float*)d_in[31];
  const float* dec_e3 = (const float*)d_in[32];
  const float* dec_w4 = (const float*)d_in[33];
  const float* dec_b4 = (const float*)d_in[34];
  const int*   sid    = (const int*)d_in[35];
  const float* tms    = (const float*)d_in[36];

  float* out    = (float*)d_out;
  float* sigmaO = out + 20971520L;     // [64,32,32]
  float* betaO  = out + 21037056L;     // [2,64]

  float* ws   = (float*)d_ws;
  float* big0 = ws;                    // 20,971,520 : enc1 out / convT3 out
  float* big1 = ws + 20971520L;        //  5,242,880 : enc2 out / convT2 out
  float* big2 = ws + 26214400L;        //  1,310,720 : enc3 out
  float* big3 = ws + 27525120L;        //    655,360 : convT1 out
  float* big4 = ws + 28180480L;        //    602,112 : enc4 out (h, [32,18816])
  float* big5 = ws + 28782592L;        //    602,112 : dec linear out
  float* zbuf = ws + 29384704L;        //      2,048 : z [32,64]
  float* zhat = ws + 29386752L;        //      2,048 : z_hat [32,64]
  float* mv   = ws + 29388800L;        //        128 : BN mean/var scratch

  auto gridc = [](long npos, int cout) {
    return dim3((unsigned)((npos + 15) / 16), (unsigned)((cout + 15) / 16), 1);
  };

  // ----------------- encoder -----------------
  conv3d_wmma<<<gridc(32L*64*64*20, 8), 32, 0, stream>>>(x, enc_w1, enc_b1, big0,
      32, 1, 128, 128, 40,  8, 64, 64, 20,  2, 1, 0, 0);
  bn_stats<<<8, 256, 0, stream>>>(big0, mv, 32, 8, 64L*64*20);
  bn_apply_lrelu<<<8192, 256, 0, stream>>>(big0, mv, enc_g1, enc_e1, 32, 8, 64L*64*20);

  conv3d_wmma<<<gridc(32L*32*32*10, 16), 32, 0, stream>>>(big0, enc_w2, enc_b2, big1,
      32, 8, 64, 64, 20,  16, 32, 32, 10,  2, 1, 0, 0);
  bn_stats<<<16, 256, 0, stream>>>(big1, mv, 32, 16, 32L*32*10);
  bn_apply_lrelu<<<4096, 256, 0, stream>>>(big1, mv, enc_g2, enc_e2, 32, 16, 32L*32*10);

  conv3d_wmma<<<gridc(32L*16*16*5, 32), 32, 0, stream>>>(big1, enc_w3, enc_b3, big2,
      32, 16, 32, 32, 10,  32, 16, 16, 5,  2, 1, 0, 0);
  bn_stats<<<32, 256, 0, stream>>>(big2, mv, 32, 32, 16L*16*5);
  bn_apply_lrelu<<<2048, 256, 0, stream>>>(big2, mv, enc_g3, enc_e3, 32, 32, 16L*16*5);

  conv3d_wmma<<<gridc(32L*14*14*3, 32), 32, 0, stream>>>(big2, enc_w4, enc_b4, big4,
      32, 32, 16, 16, 5,  32, 14, 14, 3,  1, 0, 0, 0);
  bn_stats<<<32, 256, 0, stream>>>(big4, mv, 32, 32, 14L*14*3);
  bn_apply_lrelu<<<1024, 256, 0, stream>>>(big4, mv, enc_g4, enc_e4, 32, 32, 14L*14*3);

  // mu linear
  gemm_nt_wmma<<<dim3(2, 4), 32, 0, stream>>>(mu_w, big4, mu_b, zbuf, 64, 32, 18816);

  // IGLS + eigen + sample
  igls_eig<<<64, 128, 0, stream>>>(zbuf, sid, tms, sigmaO, betaO, zhat, 64);

  // ----------------- decoder -----------------
  gemm_nt_wmma<<<dim3(2, 1176), 32, 0, stream>>>(dec_lw, zhat, dec_lb, big5, 18816, 32, 64);

  conv3d_wmma<<<gridc(32L*16*16*5, 16), 32, 0, stream>>>(big5, dec_w1, dec_b1, big3,
      32, 32, 14, 14, 3,  16, 16, 16, 5,  1, 2, 1, 0);
  bn_stats<<<16, 256, 0, stream>>>(big3, mv, 32, 16, 16L*16*5);
  bn_apply_lrelu<<<1024, 256, 0, stream>>>(big3, mv, dec_g1, dec_e1, 32, 16, 16L*16*5);

  conv3d_wmma<<<gridc(32L*32*32*10, 16), 32, 0, stream>>>(big3, dec_w2, dec_b2, big1,
      32, 16, 16, 16, 5,  16, 32, 32, 10,  2, 1, 1, 0);
  bn_stats<<<16, 256, 0, stream>>>(big1, mv, 32, 16, 32L*32*10);
  bn_apply_lrelu<<<4096, 256, 0, stream>>>(big1, mv, dec_g2, dec_e2, 32, 16, 32L*32*10);

  conv3d_wmma<<<gridc(32L*64*64*20, 8), 32, 0, stream>>>(big1, dec_w3, dec_b3, big0,
      32, 16, 32, 32, 10,  8, 64, 64, 20,  2, 1, 1, 0);
  bn_stats<<<8, 256, 0, stream>>>(big0, mv, 32, 8, 64L*64*20);
  bn_apply_lrelu<<<8192, 256, 0, stream>>>(big0, mv, dec_g3, dec_e3, 32, 8, 64L*64*20);

  conv3d_wmma<<<gridc(32L*128*128*40, 1), 32, 0, stream>>>(big0, dec_w4, dec_b4, out,
      32, 8, 64, 64, 20,  1, 128, 128, 40,  2, 1, 1, 1);
}
